// ETNN_52063593562431
// MI455X (gfx1250) — matrix operand order repaired and checked by reading, not compiled, not run on hardware
//
#include <hip/hip_runtime.h>
#include <cstddef>

// ---------------------------------------------------------------------------
// CDNA5 (gfx1250) wave32 WMMA types
// ---------------------------------------------------------------------------
typedef __attribute__((ext_vector_type(16))) _Float16 v16h;
typedef __attribute__((ext_vector_type(8)))  float    v8f;

#if defined(__has_builtin)
#if __has_builtin(__builtin_amdgcn_global_load_async_to_lds_b128) && \
    __has_builtin(__builtin_amdgcn_s_wait_asynccnt)
#define HAVE_ASYNC_LDS 1
#endif
#endif
#ifndef HAVE_ASYNC_LDS
#define HAVE_ASYNC_LDS 0
#endif

#if HAVE_ASYNC_LDS
typedef int v4i_vs __attribute__((vector_size(16)));
typedef __attribute__((address_space(1))) v4i_vs gbl_v4i;
typedef __attribute__((address_space(3))) v4i_vs lds_v4i;
#endif

__device__ __forceinline__ v8f wmma_f16(v16h a, v16h b, v8f c) {
  // D = A(16x32,f16) x B(32x16,f16) + C(16x16,f32)
  return __builtin_amdgcn_wmma_f32_16x16x32_f16(
      /*neg_a=*/false, a, /*neg_b=*/false, b,
      /*c_mod=*/(short)0, c, /*reuse_a=*/false, /*reuse_b=*/false);
}

// A-fragment: 16-bit A 16x32 in this code's contiguous k order (matches the B panel):
// lane L (M = L&15), group g = L>>4:
//   halves 0..15 -> K = c*32 + g*16 + j
__device__ __forceinline__ v16h a_frag_f32(const float* __restrict__ base, int c, int g) {
  v16h a;
  const int k0 = c * 32 + g * 16;
#pragma unroll
  for (int j = 0; j < 8; ++j) a[j] = (_Float16)base[k0 + j];
#pragma unroll
  for (int j = 0; j < 8; ++j) a[8 + j] = (_Float16)base[k0 + 8 + j];
  return a;
}

// Weight panels live pre-swizzled in global memory (f16) with the LDS layout:
// B 32x16 layout: lane L holds column n = t*16 + (L&15);
//   halves j=0..15 -> K = c*32 + (L>>4)*16 + j.
// LDS index: (((c*8 + t)*32) + L)*16 + j    (panel = 128 K x 128 N = 16384 halves)
// The per-block staging is therefore a straight contiguous 32KB copy; use the
// CDNA5 async global->LDS path when the builtins are available.
__device__ __forceinline__ void copy_panel(_Float16* sW, const _Float16* __restrict__ Wsw,
                                           int panel, int tid) {
  const _Float16* src = Wsw + (size_t)panel * 16384;
#if HAVE_ASYNC_LDS
  const char* s8 = (const char*)src;
  // low 32 bits of a flat shared pointer are the LDS byte address
  const unsigned ldsBase = (unsigned)(size_t)(void*)sW;
#pragma unroll
  for (int j = 0; j < 8; ++j) {
    const int o = (tid + j * 256) * 16;
    __builtin_amdgcn_global_load_async_to_lds_b128(
        (gbl_v4i*)(size_t)(s8 + o),
        (lds_v4i*)(size_t)(ldsBase + o),
        0, 0);
  }
  __builtin_amdgcn_s_wait_asynccnt(0);
#else
  const uint4* s4 = (const uint4*)src;
  uint4* d4 = (uint4*)sW;
#pragma unroll
  for (int j = 0; j < 8; ++j) d4[tid + j * 256] = s4[tid + j * 256];
#endif
}

__device__ __forceinline__ float dist3(float ax, float ay, float az,
                                       float bx, float by, float bz) {
  const float dx = ax - bx, dy = ay - by, dz = az - bz;
  return sqrtf(dx * dx + dy * dy + dz * dz + 1e-12f);
}

// ---------------------------------------------------------------------------
// Utility kernels
// ---------------------------------------------------------------------------
__global__ void zero_kernel(float* p, long long n) {
  long long i = (long long)blockIdx.x * blockDim.x + threadIdx.x;
  const long long stride = (long long)gridDim.x * blockDim.x;
  for (; i < n; i += stride) p[i] = 0.f;
}

// pre-swizzle a [npanels*128 x 128] f32 weight matrix into f16 LDS-fragment order
__global__ void swz_kernel(const float* __restrict__ W, _Float16* __restrict__ out,
                           int npanels) {
  const int idx = blockIdx.x * blockDim.x + threadIdx.x;
  if (idx >= npanels * 16384) return;
  const int p = idx >> 14;
  const int u = idx & 16383;
  const int j = u & 15;
  const int L = (u >> 4) & 31;
  const int t = (u >> 9) & 7;
  const int c = u >> 12;
  const int k = c * 32 + ((L >> 4) << 4) + j;
  const int n = t * 16 + (L & 15);
  out[idx] = (_Float16)W[(p * 128 + k) * 128 + n];
}

template <int K>
__global__ void cellstats_kernel(const float* __restrict__ pos, const int* __restrict__ cells,
                                 int n, float* __restrict__ cent, float* __restrict__ diam) {
  const int c = blockIdx.x * blockDim.x + threadIdx.x;
  if (c >= n) return;
  float px[K], py[K], pz[K];
  float cx = 0.f, cy = 0.f, cz = 0.f;
#pragma unroll
  for (int i = 0; i < K; ++i) {
    const int nd = cells[c * K + i];
    px[i] = pos[nd * 3 + 0]; py[i] = pos[nd * 3 + 1]; pz[i] = pos[nd * 3 + 2];
    cx += px[i]; cy += py[i]; cz += pz[i];
  }
  const float inv = 1.f / (float)K;
  cx *= inv; cy *= inv; cz *= inv;
  float dmax = 1e-6f;  // self-distance = sqrt(1e-12)
#pragma unroll
  for (int i = 0; i < K; ++i)
#pragma unroll
    for (int j = i + 1; j < K; ++j)
      dmax = fmaxf(dmax, dist3(px[i], py[i], pz[i], px[j], py[j], pz[j]));
  cent[c * 3 + 0] = cx; cent[c * 3 + 1] = cy; cent[c * 3 + 2] = cz;
  diam[c] = dmax;
}

// invariants per adjacency + fused sum/sumsq stats for batch-norm
template <int KS, int KR>
__global__ void inv_kernel(const float* __restrict__ pos,
                           const int* __restrict__ cs, const float* __restrict__ cents,
                           const float* __restrict__ diams,
                           const int* __restrict__ cr, const float* __restrict__ centr,
                           const float* __restrict__ diamr,
                           const int* __restrict__ adj, int E,
                           float* __restrict__ inv, float* __restrict__ stats) {
  __shared__ float sred[10];
  const int tid = threadIdx.x;
  if (tid < 10) sred[tid] = 0.f;
  __syncthreads();
  const int e = blockIdx.x * blockDim.x + tid;
  if (e < E) {
    const int s = adj[e], r = adj[E + e];
    float psx[KS], psy[KS], psz[KS], prx[KR], pry[KR], prz[KR];
    if constexpr (KS == 1) {
      psx[0] = pos[s * 3]; psy[0] = pos[s * 3 + 1]; psz[0] = pos[s * 3 + 2];
    } else {
#pragma unroll
      for (int i = 0; i < KS; ++i) {
        const int nd = cs[s * KS + i];
        psx[i] = pos[nd * 3]; psy[i] = pos[nd * 3 + 1]; psz[i] = pos[nd * 3 + 2];
      }
    }
    if constexpr (KR == 1) {
      prx[0] = pos[r * 3]; pry[0] = pos[r * 3 + 1]; prz[0] = pos[r * 3 + 2];
    } else {
#pragma unroll
      for (int j = 0; j < KR; ++j) {
        const int nd = cr[r * KR + j];
        prx[j] = pos[nd * 3]; pry[j] = pos[nd * 3 + 1]; prz[j] = pos[nd * 3 + 2];
      }
    }
    const float cd = dist3(cents[s * 3], cents[s * 3 + 1], cents[s * 3 + 2],
                           centr[r * 3], centr[r * 3 + 1], centr[r * 3 + 2]);
    const float ds = (KS == 1) ? 1e-6f : diams[s];
    const float dr = (KR == 1) ? 1e-6f : diamr[r];
    float colmin[KR];
#pragma unroll
    for (int j = 0; j < KR; ++j) colmin[j] = 3.4e38f;
    float hsr = 0.f;
#pragma unroll
    for (int i = 0; i < KS; ++i) {
      float rmin = 3.4e38f;
#pragma unroll
      for (int j = 0; j < KR; ++j) {
        const float d = dist3(psx[i], psy[i], psz[i], prx[j], pry[j], prz[j]);
        rmin = fminf(rmin, d);
        colmin[j] = fminf(colmin[j], d);
      }
      hsr = fmaxf(hsr, rmin);
    }
    float hrs = 0.f;
#pragma unroll
    for (int j = 0; j < KR; ++j) hrs = fmaxf(hrs, colmin[j]);
    float v[5] = {cd, ds, dr, hsr, hrs};
#pragma unroll
    for (int q = 0; q < 5; ++q) {
      inv[(size_t)e * 5 + q] = v[q];
      atomicAdd(&sred[q], v[q]);
      atomicAdd(&sred[5 + q], v[q] * v[q]);
    }
  }
  __syncthreads();
  if (tid < 10) atomicAdd(&stats[tid], sred[tid]);
}

template <int K>
__global__ void meanfeat_kernel(const float* __restrict__ x, const int* __restrict__ cells,
                                int n, float* __restrict__ out) {
  const long long idx = (long long)blockIdx.x * blockDim.x + threadIdx.x;
  if (idx >= (long long)n * 128) return;
  const int row = (int)(idx >> 7), f = (int)(idx & 127);
  float s = 0.f;
#pragma unroll
  for (int i = 0; i < K; ++i) s += x[(size_t)cells[row * K + i] * 128 + f];
  out[idx] = s * (1.f / (float)K);
}

// ---------------------------------------------------------------------------
// Message-passing kernel: m = SiLU([h_s|h_r|inv_bn] @ W + b),
// w = sigmoid(m @ infW + infb), scatter-add m*w into mes[recv].
// block = 256 threads = 8 waves; 16 edges per wave; K = 256 via WMMA + 5 tail.
// ---------------------------------------------------------------------------
__global__ __launch_bounds__(256)
void msg_kernel(const float* __restrict__ hs, const float* __restrict__ hr,
                const int* __restrict__ adj, int E,
                const float* __restrict__ inv, const float* __restrict__ stats,
                const _Float16* __restrict__ Wsw,
                const float* __restrict__ msgW, const float* __restrict__ msgB,
                const float* __restrict__ infW, const float* __restrict__ infB,
                float* __restrict__ mes) {
  __shared__ __align__(32) _Float16 sW[16384];  // 128K x 128N f16 panel, swizzled
  __shared__ float sTail[640];                  // msg_W rows 256..260
  __shared__ float sBias[128];
  __shared__ float sInf[128];
  const int tid = threadIdx.x;
  for (int u = tid; u < 640; u += 256) sTail[u] = msgW[(256 + (u >> 7)) * 128 + (u & 127)];
  if (tid < 128) { sBias[tid] = msgB[tid]; sInf[tid] = infW[tid]; }

  // batch-norm stats (training batch statistics)
  float mu[5], rs[5];
  const float invE = 1.f / (float)E;
#pragma unroll
  for (int q = 0; q < 5; ++q) {
    const float m0 = stats[q] * invE;
    const float var = stats[5 + q] * invE - m0 * m0;
    mu[q] = m0;
    rs[q] = rsqrtf(fmaxf(var, 0.f) + 1e-5f);
  }

  const int warp = tid >> 5, lane = tid & 31;
  const int g = lane >> 4, col = lane & 15;
  const int tb = blockIdx.x * 128 + warp * 16;

  // A-row gather bases (row M = lane&15)
  const int eA = tb + col;
  const int eAc = (eA < E) ? eA : (E - 1);
  const float* base0 = hs + (size_t)adj[eAc] * 128;
  const float* base1 = hr + (size_t)adj[E + eAc] * 128;

  const v8f z8 = {0.f, 0.f, 0.f, 0.f, 0.f, 0.f, 0.f, 0.f};
  v8f acc[8];
#pragma unroll
  for (int t = 0; t < 8; ++t) acc[t] = z8;

#pragma unroll
  for (int p = 0; p < 2; ++p) {
    __syncthreads();
    copy_panel(sW, Wsw, p, tid);
    __syncthreads();
    const float* ab = p ? base1 : base0;
#pragma unroll
    for (int c = 0; c < 4; ++c) {
      const v16h a = a_frag_f32(ab, c, g);
#pragma unroll
      for (int t = 0; t < 8; ++t) {
        const v16h bf = *(const v16h*)(sW + (((c * 8 + t) * 32) + lane) * 16);
        acc[t] = wmma_f16(a, bf, acc[t]);
      }
    }
  }

  // Epilogue: rows handled by this lane are M = i + 8*g (C/D layout)
  float nv[8][5];
  int rrow[8];
  bool ok[8];
#pragma unroll
  for (int i = 0; i < 8; ++i) {
    const int e = tb + i + 8 * g;
    const int ec = (e < E) ? e : (E - 1);
    ok[i] = (e < E);
    rrow[i] = adj[E + ec];
#pragma unroll
    for (int q = 0; q < 5; ++q) nv[i][q] = (inv[(size_t)ec * 5 + q] - mu[q]) * rs[q];
  }
  // bias + invariant tail + SiLU
#pragma unroll
  for (int t = 0; t < 8; ++t) {
    const int n = t * 16 + col;
    const float bn = sBias[n];
    const float t0 = sTail[n], t1 = sTail[128 + n], t2 = sTail[256 + n],
                t3 = sTail[384 + n], t4 = sTail[512 + n];
#pragma unroll
    for (int i = 0; i < 8; ++i) {
      const float xv = acc[t][i] + bn + nv[i][0] * t0 + nv[i][1] * t1 + nv[i][2] * t2 +
                       nv[i][3] * t3 + nv[i][4] * t4;
      acc[t][i] = xv / (1.f + __expf(-xv));  // SiLU
    }
  }
  // edge-inference gate: dot(m, infW) reduced across the 16-lane half
  float pi[8];
#pragma unroll
  for (int i = 0; i < 8; ++i) pi[i] = 0.f;
#pragma unroll
  for (int t = 0; t < 8; ++t) {
    const float wg = sInf[t * 16 + col];
#pragma unroll
    for (int i = 0; i < 8; ++i) pi[i] += acc[t][i] * wg;
  }
#pragma unroll
  for (int m = 1; m < 16; m <<= 1)
#pragma unroll
    for (int i = 0; i < 8; ++i) pi[i] += __shfl_xor(pi[i], m, 32);
  const float ib = infB[0];
#pragma unroll
  for (int i = 0; i < 8; ++i) pi[i] = 1.f / (1.f + __expf(-(pi[i] + ib)));
  // scatter-add gated messages
#pragma unroll
  for (int i = 0; i < 8; ++i)
    if (ok[i]) {
      float* dst = mes + (size_t)rrow[i] * 128 + col;
      const float wv = pi[i];
#pragma unroll
      for (int t = 0; t < 8; ++t) atomicAdd(dst + t * 16, acc[t][i] * wv);
    }
}

// ---------------------------------------------------------------------------
// Generic linear: out[row] = (resid?) + concat(parts)[row] @ W + b
// Used for embedding (1 part) and residual updates (2-3 parts, in-place).
// ---------------------------------------------------------------------------
__global__ __launch_bounds__(256)
void lin_kernel(const float* __restrict__ s0, const float* __restrict__ s1,
                const float* __restrict__ s2, int nparts, int nrows,
                const _Float16* __restrict__ Wsw, const float* __restrict__ bias,
                const float* __restrict__ resid, float* __restrict__ out) {
  __shared__ __align__(32) _Float16 sW[16384];
  __shared__ float sBias[128];
  const int tid = threadIdx.x;
  if (tid < 128) sBias[tid] = bias[tid];
  const int warp = tid >> 5, lane = tid & 31;
  const int g = lane >> 4, col = lane & 15;
  const int tb = blockIdx.x * 128 + warp * 16;
  int rA = tb + col;
  if (rA >= nrows) rA = nrows - 1;
  const float* bases[3];
  bases[0] = s0 + (size_t)rA * 128;
  bases[1] = s1 ? (s1 + (size_t)rA * 128) : bases[0];
  bases[2] = s2 ? (s2 + (size_t)rA * 128) : bases[0];

  const v8f z8 = {0.f, 0.f, 0.f, 0.f, 0.f, 0.f, 0.f, 0.f};
  v8f acc[8];
#pragma unroll
  for (int t = 0; t < 8; ++t) acc[t] = z8;

  for (int p = 0; p < nparts; ++p) {
    __syncthreads();
    copy_panel(sW, Wsw, p, tid);
    __syncthreads();
    const float* ab = bases[p];
#pragma unroll
    for (int c = 0; c < 4; ++c) {
      const v16h a = a_frag_f32(ab, c, g);
#pragma unroll
      for (int t = 0; t < 8; ++t) {
        const v16h bf = *(const v16h*)(sW + (((c * 8 + t) * 32) + lane) * 16);
        acc[t] = wmma_f16(a, bf, acc[t]);
      }
    }
  }

#pragma unroll
  for (int t = 0; t < 8; ++t) {
    const int n = t * 16 + col;
    const float bn = sBias[n];
#pragma unroll
    for (int i = 0; i < 8; ++i) {
      const int row = tb + i + 8 * g;
      if (row < nrows) {
        const size_t o = (size_t)row * 128 + n;
        float v = acc[t][i] + bn;
        if (resid) v += resid[o];
        out[o] = v;
      }
    }
  }
}

// head: out[row] = h0[row,:] @ preW + preb ; d_out = [out, out, 0]
__global__ void head_kernel(const float* __restrict__ h0, const float* __restrict__ W,
                            const float* __restrict__ b, float* __restrict__ out, int N) {
  const int gid = blockIdx.x * blockDim.x + threadIdx.x;
  const int row = gid >> 5;
  const int lane = threadIdx.x & 31;
  if (row >= N) return;
  const float* hrow = h0 + (size_t)row * 128;
  float a = 0.f;
#pragma unroll
  for (int j = 0; j < 4; ++j) {
    const int f = lane + j * 32;
    a += hrow[f] * W[f];
  }
#pragma unroll
  for (int m = 16; m >= 1; m >>= 1) a += __shfl_xor(a, m, 32);
  if (lane == 0) {
    const float v = a + b[0];
    out[row] = v;
    out[N + row] = v;
    if (row == 0) out[2 * N] = 0.f;  // x_sub_feat
  }
}

// ---------------------------------------------------------------------------
// Host launcher
// ---------------------------------------------------------------------------
extern "C" void kernel_launch(void* const* d_in, const int* in_sizes, int n_in,
                              void* d_out, int out_size, void* d_ws, size_t ws_size,
                              hipStream_t stream) {
  (void)n_in; (void)out_size; (void)ws_size;
  const float* pos = (const float*)d_in[0];
  const float* x   = (const float*)d_in[1];
  const int* cell1 = (const int*)d_in[2];
  const int* cell2 = (const int*)d_in[3];
  const int* adj[4] = {(const int*)d_in[4], (const int*)d_in[5],
                       (const int*)d_in[6], (const int*)d_in[7]};
  const int N  = in_sizes[0] / 3;
  const int E1 = in_sizes[2] / 2;
  const int E2 = in_sizes[3] / 6;
  const int A[4] = {in_sizes[4] / 2, in_sizes[5] / 2, in_sizes[6] / 2, in_sizes[7] / 2};

  auto P = [&](int i) { return (const float*)d_in[i]; };
  // params flattened via jax tree_leaves (sorted dict keys): emb, layers, pre
  const float* embW[3];
  const float* embB[3];
  for (int d = 0; d < 3; ++d) { embW[d] = P(8 + 2 * d); embB[d] = P(8 + 2 * d + 1); }
  // per layer (22 leaves): 4 x [inf_W, inf_b, msg_W, msg_b], then 3 x [upd_W, upd_b]
  const int LB = 14;
  const float* preW0 = P(LB + 88 + 0);
  const float* preB0 = P(LB + 88 + 1);

  // workspace layout (floats)
  float* w = (float*)d_ws;
  size_t off = 0;
  float* h0 = w + off;      off += (size_t)N * 128;
  float* h1 = w + off;      off += (size_t)E1 * 128;
  float* h2 = w + off;      off += (size_t)E2 * 128;
  float* mes00 = w + off;   off += (size_t)N * 128;
  float* mes01 = w + off;   off += (size_t)E1 * 128;   // also feats1 before layer 0
  float* mes11 = w + off;   off += (size_t)E1 * 128;
  float* mes12 = w + off;   off += (size_t)E2 * 128;   // also feats2 before layer 0
  float* invb[4];
  for (int a = 0; a < 4; ++a) { invb[a] = w + off; off += (size_t)A[a] * 5; }
  float* cent1 = w + off;   off += (size_t)E1 * 3;
  float* diam1 = w + off;   off += (size_t)E1;
  float* cent2 = w + off;   off += (size_t)E2 * 3;
  float* diam2 = w + off;   off += (size_t)E2;
  float* stats = w + off;   off += 40;
  off = (off + 3) & ~(size_t)3;                // 16-byte align the f16 region
  _Float16* wsw = (_Float16*)(w + off);

  const int T = 256;
  auto cdiv = [](long long a, long long b) { return (unsigned)((a + b - 1) / b); };

  // ---- pre-swizzle all weight matrices into f16 LDS-fragment order (once) ----
  size_t off16 = 0;
  auto swz = [&](const float* W, int npanels) {
    _Float16* dst = wsw + off16;
    swz_kernel<<<cdiv((long long)npanels * 16384, T), T, 0, stream>>>(W, dst, npanels);
    off16 += (size_t)npanels * 16384;
    return (const _Float16*)dst;
  };
  const _Float16* embS[3];
  for (int d = 0; d < 3; ++d) embS[d] = swz(embW[d], 1);
  const _Float16* msgS[4][4];
  const _Float16* updS[4][3];
  const int uparts[3] = {2, 3, 2};
  for (int l = 0; l < 4; ++l) {
    const int lb = LB + 22 * l;
    for (int a = 0; a < 4; ++a) msgS[l][a] = swz(P(lb + 4 * a + 2), 2);
    const int ub = lb + 16;
    for (int d = 0; d < 3; ++d) updS[l][d] = swz(P(ub + 2 * d), uparts[d]);
  }

  // invariant stats buffers
  zero_kernel<<<1, 64, 0, stream>>>(stats, 40);
  // cell centroids / diameters
  cellstats_kernel<2><<<cdiv(E1, T), T, 0, stream>>>(pos, cell1, E1, cent1, diam1);
  cellstats_kernel<6><<<cdiv(E2, T), T, 0, stream>>>(pos, cell2, E2, cent2, diam2);
  // invariants (+ fused BN statistics)
  inv_kernel<1, 1><<<cdiv(A[0], T), T, 0, stream>>>(pos, nullptr, pos, nullptr,
                                                    nullptr, pos, nullptr,
                                                    adj[0], A[0], invb[0], stats + 0);
  inv_kernel<1, 2><<<cdiv(A[1], T), T, 0, stream>>>(pos, nullptr, pos, nullptr,
                                                    cell1, cent1, diam1,
                                                    adj[1], A[1], invb[1], stats + 10);
  inv_kernel<2, 2><<<cdiv(A[2], T), T, 0, stream>>>(pos, cell1, cent1, diam1,
                                                    cell1, cent1, diam1,
                                                    adj[2], A[2], invb[2], stats + 20);
  inv_kernel<2, 6><<<cdiv(A[3], T), T, 0, stream>>>(pos, cell1, cent1, diam1,
                                                    cell2, cent2, diam2,
                                                    adj[3], A[3], invb[3], stats + 30);
  // initial features (means), staged into mes01 / mes12 before the layer loop
  meanfeat_kernel<2><<<cdiv((long long)E1 * 128, T), T, 0, stream>>>(x, cell1, E1, mes01);
  meanfeat_kernel<6><<<cdiv((long long)E2 * 128, T), T, 0, stream>>>(x, cell2, E2, mes12);
  // embeddings: h[d] = feats[d] @ embW + embB
  lin_kernel<<<cdiv(N, 128), T, 0, stream>>>(x, nullptr, nullptr, 1, N, embS[0], embB[0],
                                             nullptr, h0);
  lin_kernel<<<cdiv(E1, 128), T, 0, stream>>>(mes01, nullptr, nullptr, 1, E1, embS[1],
                                              embB[1], nullptr, h1);
  lin_kernel<<<cdiv(E2, 128), T, 0, stream>>>(mes12, nullptr, nullptr, 1, E2, embS[2],
                                              embB[2], nullptr, h2);

  const float* HS[4] = {h0, h0, h1, h1};
  const float* HR[4] = {h0, h1, h1, h2};
  float* MES[4] = {mes00, mes01, mes11, mes12};
  const long long mlen[4] = {(long long)N * 128, (long long)E1 * 128,
                             (long long)E1 * 128, (long long)E2 * 128};

  for (int l = 0; l < 4; ++l) {
    const int lb = LB + 22 * l;
    for (int a = 0; a < 4; ++a)
      zero_kernel<<<2048, T, 0, stream>>>(MES[a], mlen[a]);
    for (int a = 0; a < 4; ++a) {
      const float* infW = P(lb + 4 * a + 0);
      const float* infB = P(lb + 4 * a + 1);
      const float* msgW = P(lb + 4 * a + 2);
      const float* msgB = P(lb + 4 * a + 3);
      msg_kernel<<<cdiv(A[a], 128), T, 0, stream>>>(HS[a], HR[a], adj[a], A[a], invb[a],
                                                    stats + 10 * a, msgS[l][a], msgW, msgB,
                                                    infW, infB, MES[a]);
    }
    const int ub = lb + 16;
    // dim 0: [h0 | mes00] -> h0 (residual, in-place)
    lin_kernel<<<cdiv(N, 128), T, 0, stream>>>(h0, mes00, nullptr, 2, N, updS[l][0],
                                               P(ub + 1), h0, h0);
    // dim 1: [h1 | mes01 | mes11] -> h1
    lin_kernel<<<cdiv(E1, 128), T, 0, stream>>>(h1, mes01, mes11, 3, E1, updS[l][1],
                                                P(ub + 3), h1, h1);
    // dim 2: [h2 | mes12] -> h2
    lin_kernel<<<cdiv(E2, 128), T, 0, stream>>>(h2, mes12, nullptr, 2, E2, updS[l][2],
                                                P(ub + 5), h2, h2);
  }

  // readout on dim-0 features; output = [out, out, x_sub_feat(=0)]
  head_kernel<<<cdiv((long long)N * 32, T), T, 0, stream>>>(h0, preW0, preB0,
                                                            (float*)d_out, N);
}
